// PyTorchINT4Linear_68951404970062
// MI455X (gfx1250) — matrix operand order repaired and checked by reading, not compiled, not run on hardware
//
#include <hip/hip_runtime.h>

typedef _Float16 h1;
typedef __attribute__((ext_vector_type(2)))  _Float16 h2;
typedef __attribute__((ext_vector_type(8)))  _Float16 h8;
typedef __attribute__((ext_vector_type(16))) _Float16 v16h;
typedef __attribute__((ext_vector_type(8)))  float    v8f;
typedef __attribute__((ext_vector_type(4)))  int      i4v;

namespace {
constexpr int kInF   = 4096;
constexpr int kOutF  = 11008;
constexpr int kGroup = 128;
constexpr int kPk    = kOutF / 2;   // packed int32 per k-row
constexpr int KC     = 128;         // K per staged chunk == one quant group
constexpr int NT     = 32;          // output columns per WG (2 waves x 16)
constexpr int JT     = 16;          // packed columns per WG
constexpr int ROWH   = 136;         // padded halves per LDS row (68 dw -> stride 4 banks)
constexpr int NCH    = kInF / KC;   // 32 K-chunks
}

static __device__ __forceinline__ h2 bits_to_h2(unsigned u) {
  union { unsigned u; h2 h; } c; c.u = u; return c.h;
}

__global__ __launch_bounds__(64)
void int4lin_wmma(const h1* __restrict__ x, const int* __restrict__ wq,
                  const h1* __restrict__ ws, const h1* __restrict__ wz,
                  float* __restrict__ out)
{
  // Double-buffered dequantized-B tile [n=32][k=128] and A tile [m=64][k=128].
  __shared__ _Float16 ldsB[2][NT * ROWH];
  __shared__ _Float16 ldsA[2][64 * ROWH];

  const int tid   = threadIdx.x;
  const int lane  = tid & 31;
  const int wave  = tid >> 5;               // 0..1
  const int jbase = blockIdx.x * JT;        // packed-col base
  const int nbase = blockIdx.x * NT;        // out-col base

  // Weight loader: thread owns 4 packed cols x 8 CONSECUTIVE k-rows.
  const int ljj = (tid & 3) * 4;            // local packed col 0..12
  const int lr8 = (tid >> 2) * 8;           // k-row base 0..120
  const int am  = tid;                      // A loader: one 128-half row

  // Compute mapping (wave32; contiguous k = 16*(lane/16) + s in both A and B).
  const int cn   = wave * 16 + (lane & 15); // local out col owned by lane
  const int ksel = (lane >> 4) * 16;        // A frag K select
  const int khi  = (lane >> 4) * 16;        // B frag K select

  v8f acc[4] = {};
  i4v wreg[8];
  h8  s8, z8;

  auto load_w = [&](int c) {
    const int k0 = c * KC;
    const int* p = wq + (size_t)(k0 + lr8) * kPk + jbase + ljj;
#pragma unroll
    for (int i = 0; i < 8; ++i)
      wreg[i] = *(const i4v*)(p + (size_t)i * kPk);
    // chunk == quant group: scales/zeros loaded exactly once per chunk
    s8 = *(const h8*)(ws + (size_t)c * kOutF + nbase + 2 * ljj);
    z8 = *(const h8*)(wz + (size_t)c * kOutF + nbase + 2 * ljj);
  };

  // gfx1250 async DMA: 256B x-row slice -> LDS, tracked by ASYNCcnt.
  auto load_a_async = [&](int c, int buf) {
    const h1* xp = x + (size_t)am * kInF + c * KC;      // 256B, 16B-aligned
    unsigned la = (unsigned)(uintptr_t)&ldsA[buf][(size_t)am * ROWH];
    asm volatile(
        "global_load_async_to_lds_b128 %0, %1, off\n\t"
        "global_load_async_to_lds_b128 %0, %1, off offset:16\n\t"
        "global_load_async_to_lds_b128 %0, %1, off offset:32\n\t"
        "global_load_async_to_lds_b128 %0, %1, off offset:48\n\t"
        "global_load_async_to_lds_b128 %0, %1, off offset:64\n\t"
        "global_load_async_to_lds_b128 %0, %1, off offset:80\n\t"
        "global_load_async_to_lds_b128 %0, %1, off offset:96\n\t"
        "global_load_async_to_lds_b128 %0, %1, off offset:112\n\t"
        "global_load_async_to_lds_b128 %0, %1, off offset:128\n\t"
        "global_load_async_to_lds_b128 %0, %1, off offset:144\n\t"
        "global_load_async_to_lds_b128 %0, %1, off offset:160\n\t"
        "global_load_async_to_lds_b128 %0, %1, off offset:176\n\t"
        "global_load_async_to_lds_b128 %0, %1, off offset:192\n\t"
        "global_load_async_to_lds_b128 %0, %1, off offset:208\n\t"
        "global_load_async_to_lds_b128 %0, %1, off offset:224\n\t"
        "global_load_async_to_lds_b128 %0, %1, off offset:240"
        :: "v"(la), "v"(xp) : "memory");
  };

  // int4 -> fp16 via mantissa-OR: 0x6400|n == half(1024+n); (q-z)*s folds into
  // one v_pk_fma_f16 with bias -(z+1024)*s.  Upper 24 bits of each packed
  // int32 are zero, so two rows merge with one lshl_or; per row-pair:
  // lshl_or + lshr + 2x and_or + 2x pk_fma for 4 weights.
  auto stage_w = [&](int buf) {
    const h1 kK = (h1)1024.0f;
    const h8 k1024 = {kK, kK, kK, kK, kK, kK, kK, kK};
    const h8 badj = -((z8 + k1024) * s8);
    _Float16* bb = &ldsB[buf][0];
#pragma unroll
    for (int q = 0; q < 4; ++q) {
      const h2 se = {s8[2 * q], s8[2 * q]};
      const h2 be = {badj[2 * q], badj[2 * q]};
      const h2 so = {s8[2 * q + 1], s8[2 * q + 1]};
      const h2 bo = {badj[2 * q + 1], badj[2 * q + 1]};
      union { h8 v; h2 h[4]; } we, wo;
#pragma unroll
      for (int r = 0; r < 4; ++r) {     // k rows (lr8+2r, lr8+2r+1)
        const unsigned pa = (unsigned)wreg[2 * r][q];
        const unsigned pb = (unsigned)wreg[2 * r + 1][q];
        const unsigned t  = pa | (pb << 16);
        const unsigned lo = (t & 0x000F000Fu) | 0x64006400u;
        const unsigned hi = ((t >> 4) & 0x000F000Fu) | 0x64006400u;
        we.h[r] = bits_to_h2(lo) * se + be;   // even col
        wo.h[r] = bits_to_h2(hi) * so + bo;   // odd col
      }
      const int ne = 2 * (ljj + q);
      *(h8*)(bb + (size_t)ne * ROWH + lr8)       = we.v;   // ds_store_b128
      *(h8*)(bb + (size_t)(ne + 1) * ROWH + lr8) = wo.v;
    }
  };

  load_w(0);
  load_a_async(0, 0);
  stage_w(0);
  asm volatile("s_wait_asynccnt 0x0" ::: "memory");
  __syncthreads();

  for (int c = 0; c < NCH; ++c) {
    const int cur = c & 1;
    if (c + 1 < NCH) {
      load_w(c + 1);
      load_a_async(c + 1, cur ^ 1);
    }
    if (c + 2 < NCH)
      __builtin_prefetch(wq + (size_t)((c + 2) * KC + lr8) * kPk + jbase + ljj, 0, 0);

    {  // WMMA on current LDS buffers: 4 K-steps x 4 M-tiles
      const _Float16* bb = &ldsB[cur][0];
      const _Float16* ab = &ldsA[cur][0];
#pragma unroll
      for (int kstep = 0; kstep < 4; ++kstep) {
        union { v16h v; h8 h[2]; } B;
        const _Float16* br = bb + (size_t)cn * ROWH + kstep * 32 + khi;
        B.h[0] = *(const h8*)(br);
        B.h[1] = *(const h8*)(br + 8);
#pragma unroll
        for (int mt = 0; mt < 4; ++mt) {
          union { v16h v; h8 h[2]; } A;
          const _Float16* ar =
              ab + (size_t)(mt * 16 + (lane & 15)) * ROWH + kstep * 32 + ksel;
          A.h[0] = *(const h8*)(ar);
          A.h[1] = *(const h8*)(ar + 8);
          acc[mt] = __builtin_amdgcn_wmma_f32_16x16x32_f16(
              false, A.v, false, B.v, (short)0, acc[mt], false, false);
        }
      }
    }

    if (c + 1 < NCH) stage_w(cur ^ 1);
    asm volatile("s_wait_asynccnt 0x0" ::: "memory");
    __syncthreads();
  }

  // C/D layout: lane {0-15,16-31} -> N = lane%16, M = vgpr + 8*(lane/16).
  const int col = nbase + cn;
#pragma unroll
  for (int mt = 0; mt < 4; ++mt) {
    const int rbase = mt * 16 + (lane >> 4) * 8;
#pragma unroll
    for (int r = 0; r < 8; ++r)
      out[(size_t)(rbase + r) * kOutF + col] = acc[mt][r];
  }
}

extern "C" void kernel_launch(void* const* d_in, const int* in_sizes, int n_in,
                              void* d_out, int out_size, void* d_ws, size_t ws_size,
                              hipStream_t stream) {
  (void)in_sizes; (void)n_in; (void)d_ws; (void)ws_size; (void)out_size;
  const h1*  x      = (const h1*)d_in[0];
  const int* wq     = (const int*)d_in[1];
  const h1*  wscale = (const h1*)d_in[2];
  const h1*  wzero  = (const h1*)d_in[3];
  float* out = (float*)d_out;
  dim3 grid(kOutF / NT);   // 344 workgroups, 2 waves each
  int4lin_wmma<<<grid, 64, 0, stream>>>(x, wq, wscale, wzero, out);
}
